// Attention_77730318123361
// MI455X (gfx1250) — compile-verified, matrix operand order checked by reading, not run on hardware
//
#include <hip/hip_runtime.h>
#include <hip/hip_bf16.h>

// ---------------- problem constants ----------------
#define DIMX 4096
#define NH   32
#define NKV  8
#define HDIM 128
#define BB   4
#define LLEN 1024
#define MROW (BB*LLEN)              // 4096
#define SCALEF 0.08838834764831845f // 1/sqrt(128)
#define LN_ROPE_BASE 9.210340371976184f

typedef __attribute__((ext_vector_type(16))) __bf16       v16bf;
typedef __attribute__((ext_vector_type(8)))  float        v8f;
typedef __attribute__((ext_vector_type(4))) unsigned int  u32x4;
typedef __attribute__((ext_vector_type(8)))  int          i32x8;
typedef __attribute__((ext_vector_type(4)))  int          i32x4;

typedef __attribute__((address_space(3))) const __bf16 lds_cbf16;
typedef __attribute__((address_space(3))) const u32x4  lds_cu32x4;

// Load one 16x32 bf16 A-fragment (or 32x16 B-fragment) lane-slice:
// lane half h selects K-offsets {8h..8h+7} and {16+8h..23+8h} (ISA 7.12.2 layout).
static __device__ __forceinline__ v16bf ldfrag(const __bf16* p, int h) {
  v16bf r;
  ((u32x4*)&r)[0] = *(const u32x4*)(p + 8*h);
  ((u32x4*)&r)[1] = *(const u32x4*)(p + 16 + 8*h);
  return r;
}

// Same, but from LDS via an addrspace(3) pointer -> ds_load_b128 (not flat).
static __device__ __forceinline__ v16bf ldfrag_lds(lds_cbf16* p, int h) {
  v16bf r;
  ((u32x4*)&r)[0] = *(lds_cu32x4*)(p + 8*h);
  ((u32x4*)&r)[1] = *(lds_cu32x4*)(p + 16 + 8*h);
  return r;
}

static __device__ __forceinline__ v8f wmma_bf16(v16bf a, v16bf b, v8f c) {
  return __builtin_amdgcn_wmma_f32_16x16x32_bf16(false, a, false, b, (short)0, c, false, false);
}

// ---------------- Tensor Data Mover: 2D bf16 tile -> LDS ----------------
// D# per CDNA5 ISA. data_size=1 (2B elements).
// LDS padding: every 16 DWORDs (32 elems / 64B) insert 4 DWORDs (8 elems / 16B)
//   -> pad_interval=3, pad_amount=3 -> LDS row stride = 40 elements.
// 6-arg builtin on this toolchain: (u32x4, i32x8, i32x4, i32x4, i32x8, cpol).
static __device__ __forceinline__ void tdm_load_2d(const __bf16* gptr, unsigned lds_byte_off,
                                                   unsigned tcols, unsigned trows,
                                                   unsigned tile_cols, unsigned tile_rows) {
  unsigned long long ga = (unsigned long long)(size_t)gptr;
  u32x4 g0;
  g0[0] = 1u;                                                  // count=1, user D#
  g0[1] = lds_byte_off;                                        // lds_addr
  g0[2] = (unsigned)(ga & 0xFFFFFFFFu);                        // global_addr lo
  g0[3] = (unsigned)((ga >> 32) & 0x01FFFFFFu) | (2u << 30);   // global_addr hi | type=2
  i32x8 g1;
  g1[0] = (int)((1u << 16) | (1u << 20) | (3u << 22) | (3u << 25)); // data_size=2B, pad 16dw->+4dw
  g1[1] = (int)((tcols & 0xFFFFu) << 16);                      // tensor_dim0[15:0]
  g1[2] = (int)(((tcols >> 16) & 0xFFFFu) | ((trows & 0xFFFFu) << 16));   // dim0 hi | dim1 lo
  g1[3] = (int)(((trows >> 16) & 0xFFFFu) | (tile_cols << 16));           // dim1 hi | tile_dim0
  g1[4] = (int)(tile_rows & 0xFFFFu);                          // tile_dim1 (tile_dim2=0)
  g1[5] = (int)tcols;                                          // tensor_dim0_stride lo
  g1[6] = 0;                                                   // stride hi | dim1_stride lo
  g1[7] = 0;
  i32x4 gz4 = {0, 0, 0, 0};
  i32x8 gz8 = {0, 0, 0, 0, 0, 0, 0, 0};
  __builtin_amdgcn_tensor_load_to_lds(g0, g1, gz4, gz4, gz8, 0);
}

// ---------------- elementwise converts ----------------
__global__ void cvt_f32_bf16(const float* __restrict__ src, __bf16* __restrict__ dst, int n) {
  int i = blockIdx.x * blockDim.x + threadIdx.x;
  if (i < n) dst[i] = (__bf16)src[i];
}

// src: [K][N] f32 row-major  ->  dst: [N][K] bf16 row-major (transposed)
__global__ void cvt_transpose_bf16(const float* __restrict__ src, __bf16* __restrict__ dst,
                                   int K, int N) {
  int n = blockIdx.x * blockDim.x + threadIdx.x;
  int k = blockIdx.y * blockDim.y + threadIdx.y;
  if (n < N && k < K) dst[(size_t)n * K + k] = (__bf16)src[(size_t)k * N + n];
}

// ---------------- bf16 WMMA GEMM: C[M,N] = A[M,K] * Bt[N,K]^T ----------------
// 256 thr (8 waves), C tile 128x128, K-step 32. TDM stages A/B tiles into
// double-buffered LDS; one barrier per K-step; waves only ds_load + WMMA.
__global__ __launch_bounds__(256) void gemm_bf16_wmma(const __bf16* __restrict__ A,
                                                      const __bf16* __restrict__ Bt,
                                                      float* __restrict__ C,
                                                      int M, int N, int K) {
  __shared__ __bf16 As[2][128][40];    // written by TDM (padded rows: 40 elems)
  __shared__ __bf16 Bs[2][128][40];

  const int tid  = threadIdx.x;
  const int lane = tid & 31;
  const int wid  = tid >> 5;
  const int h    = lane >> 4;
  const int m16  = lane & 15;
  const int wm   = wid >> 1;           // 0..3 -> 32 rows each
  const int wn   = wid & 1;            // 0..1 -> 64 cols each
  const int blockM = blockIdx.y * 128;
  const int blockN = blockIdx.x * 128;

  // LDS byte offsets (generic shared ptr low 32 bits) for TDM descriptors.
  const unsigned aoff = (unsigned)(size_t)&As[0][0][0];
  const unsigned boff = (unsigned)(size_t)&Bs[0][0][0];
  const unsigned bufbytes = 128u * 40u * 2u;   // 10240 B per buffer

  // Launder the *offsets* (scalar), then rebuild addrspace(3) pointers: the
  // TDM writes are invisible to the compiler, so the ds_loads must be opaque,
  // but we keep LDS provenance so they lower to ds_load_b128 (not flat).
  unsigned aoff_l = aoff, boff_l = boff;
  asm volatile("" : "+s"(aoff_l), "+s"(boff_l));
  lds_cbf16* asp = (lds_cbf16*)(size_t)aoff_l;
  lds_cbf16* bsp = (lds_cbf16*)(size_t)boff_l;

  v8f acc[2][4] = {};
  const int NKT = K >> 5;

  if (wid == 0) {
    tdm_load_2d(A  + (size_t)blockM * K, aoff, (unsigned)K, (unsigned)M, 32u, 128u);
    tdm_load_2d(Bt + (size_t)blockN * K, boff, (unsigned)K, (unsigned)N, 32u, 128u);
  }

  for (int kti = 0; kti < NKT; ++kti) {
    const int buf = kti & 1;
    __builtin_amdgcn_s_wait_tensorcnt(0);    // SOPP; no-op for waves with TENSORcnt==0
    __syncthreads();                         // all waves see buf's tiles; prev compute done
    if (wid == 0 && kti + 1 < NKT) {         // DMA next K-tile into the other buffer
      const size_t kadv = (size_t)(kti + 1) * 32;
      tdm_load_2d(A  + (size_t)blockM * K + kadv, aoff + (unsigned)(buf ^ 1) * bufbytes,
                  (unsigned)K, (unsigned)M, 32u, 128u);
      tdm_load_2d(Bt + (size_t)blockN * K + kadv, boff + (unsigned)(buf ^ 1) * bufbytes,
                  (unsigned)K, (unsigned)N, 32u, 128u);
    }
    lds_cbf16* ab = asp + (size_t)buf * (128 * 40);
    lds_cbf16* bb = bsp + (size_t)buf * (128 * 40);
    v16bf af[2], bf[4];
#pragma unroll
    for (int mt = 0; mt < 2; ++mt) af[mt] = ldfrag_lds(ab + (wm*32 + mt*16 + m16) * 40, h);
#pragma unroll
    for (int nt = 0; nt < 4; ++nt) bf[nt] = ldfrag_lds(bb + (wn*64 + nt*16 + m16) * 40, h);
#pragma unroll
    for (int mt = 0; mt < 2; ++mt)
#pragma unroll
      for (int nt = 0; nt < 4; ++nt)
        acc[mt][nt] = wmma_bf16(af[mt], bf[nt], acc[mt][nt]);
  }

#pragma unroll
  for (int mt = 0; mt < 2; ++mt)
#pragma unroll
    for (int nt = 0; nt < 4; ++nt)
#pragma unroll
      for (int r = 0; r < 8; ++r) {
        int row = blockM + wm*32 + mt*16 + r + 8*h;   // C-layout: lanes16-31 hold M+8
        int col = blockN + wn*64 + nt*16 + m16;
        C[(size_t)row * N + col] = acc[mt][nt][r];
      }
}

// ---------------- RoPE + relayout: raw[B*L, Hn*HD] f32 -> out[B,Hn,L,HD] bf16 -------------
__global__ void rope_relayout(const float* __restrict__ raw, __bf16* __restrict__ out,
                              int Hn, int do_rope) {
  int idx = blockIdx.x * blockDim.x + threadIdx.x;   // one thread per (b,h,l,pair)
  int total = BB * Hn * LLEN * (HDIM / 2);
  if (idx >= total) return;
  int j = idx & (HDIM/2 - 1); int t = idx >> 6;
  int l = t & (LLEN - 1);     t >>= 10;
  int hh = t % Hn;            int bb = t / Hn;
  const float* src = raw + ((size_t)(bb * LLEN + l)) * (Hn * HDIM) + hh * HDIM + 2*j;
  float x1 = src[0], x2 = src[1];
  float r1 = x1, r2 = x2;
  if (do_rope) {
    float inv = __expf(-(float)(2*j) * (LN_ROPE_BASE / (float)HDIM));
    float ang = (float)l * inv;
    float s, c; __sincosf(ang, &s, &c);
    r1 = x1 * c - x2 * s;
    r2 = x1 * s + x2 * c;
  }
  __bf16* dst = out + (((size_t)(bb * Hn + hh) * LLEN + l) * HDIM) + 2*j;
  dst[0] = (__bf16)r1;
  dst[1] = (__bf16)r2;
}

// V relayout transposed: raw[B*L, NKV*HD] f32 -> out[B,NKV,HD,L] bf16 (keys contiguous)
__global__ void v_relayout_t(const float* __restrict__ raw, __bf16* __restrict__ out) {
  int idx = blockIdx.x * blockDim.x + threadIdx.x;
  int total = BB * NKV * LLEN * HDIM;
  if (idx >= total) return;
  int d = idx & (HDIM - 1); int t = idx >> 7;
  int l = t & (LLEN - 1);   t >>= 10;
  int hh = t & (NKV - 1);   int bb = t >> 3;
  out[((size_t)(bb * NKV + hh) * HDIM + d) * LLEN + l] =
      (__bf16)raw[((size_t)(bb * LLEN + l)) * (NKV * HDIM) + hh * HDIM + d];
}

// ---------------- flash attention (causal, GQA) ----------------
// grid(L/128, NH, B), block 256 (8 waves). Each wave owns 16 q-rows, streams 32-key blocks.
__global__ __launch_bounds__(256) void flash_attn_wmma(const __bf16* __restrict__ Qb,
                                                       const __bf16* __restrict__ Kb,
                                                       const __bf16* __restrict__ Vt,
                                                       __bf16* __restrict__ Ob) {
  __shared__ __bf16 Pl[8][16][40];   // per-wave P tile (16 q x 32 k), C-layout -> A-layout bridge

  const int tid  = threadIdx.x;
  const int lane = tid & 31;
  const int wid  = tid >> 5;
  const int h    = lane >> 4;
  const int n16  = lane & 15;
  const int b    = blockIdx.z;
  const int hq   = blockIdx.y;
  const int hkv  = hq >> 2;                 // NH/NKV = 4
  const int q0   = blockIdx.x * 128 + wid * 16;

  const __bf16* Qp = Qb + ((size_t)(b * NH  + hq ) * LLEN + q0) * HDIM;
  const __bf16* Kp = Kb +  (size_t)(b * NKV + hkv) * LLEN * HDIM;
  const __bf16* Vp = Vt +  (size_t)(b * NKV + hkv) * HDIM * LLEN;

  // Q A-fragments: 16 rows x 128 (4 chunks of K=32), loaded once.
  v16bf qf[4];
#pragma unroll
  for (int c = 0; c < 4; ++c) qf[c] = ldfrag(Qp + (size_t)n16 * HDIM + c*32, h);

  float mrow[8], lrow[8], alpha[8];
  v8f ao[8] = {};
#pragma unroll
  for (int r = 0; r < 8; ++r) { mrow[r] = -3.0e38f; lrow[r] = 0.0f; }

  const int nkb = (q0 + 15) / 32 + 1;       // causal: need keys <= q0+15
  for (int kb = 0; kb < nkb; ++kb) {
    const int kbase = kb * 32;
    // S = Q * K^T  (two 16x16 score tiles for 32 keys)
    v8f s0 = {}, s1 = {};
#pragma unroll
    for (int c = 0; c < 4; ++c) {
      v16bf k0 = ldfrag(Kp + (size_t)(kbase      + n16) * HDIM + c*32, h);
      v16bf k1 = ldfrag(Kp + (size_t)(kbase + 16 + n16) * HDIM + c*32, h);
      s0 = wmma_bf16(qf[c], k0, s0);
      s1 = wmma_bf16(qf[c], k1, s1);
    }
    // online softmax per row (rows live across acc regs; cols across 16-lane halves)
#pragma unroll
    for (int r = 0; r < 8; ++r) {
      int rowg = q0 + r + 8*h;
      int c0 = kbase + n16, c1 = kbase + 16 + n16;
      float x0 = s0[r] * SCALEF + (c0 <= rowg ? 0.0f : -1.0e9f);
      float x1 = s1[r] * SCALEF + (c1 <= rowg ? 0.0f : -1.0e9f);
      float mx = fmaxf(x0, x1);
      mx = fmaxf(mx, __shfl_xor(mx, 1, 32));
      mx = fmaxf(mx, __shfl_xor(mx, 2, 32));
      mx = fmaxf(mx, __shfl_xor(mx, 4, 32));
      mx = fmaxf(mx, __shfl_xor(mx, 8, 32));
      float mn = fmaxf(mrow[r], mx);
      float a  = __expf(mrow[r] - mn);
      float p0 = __expf(x0 - mn);
      float p1 = __expf(x1 - mn);
      float ps = p0 + p1;
      ps += __shfl_xor(ps, 1, 32);
      ps += __shfl_xor(ps, 2, 32);
      ps += __shfl_xor(ps, 4, 32);
      ps += __shfl_xor(ps, 8, 32);
      lrow[r] = lrow[r] * a + ps;
      mrow[r] = mn;
      alpha[r] = a;
      int prow = r + 8*h;
      Pl[wid][prow][n16]      = (__bf16)p0;
      Pl[wid][prow][16 + n16] = (__bf16)p1;
    }
    asm volatile("s_wait_dscnt 0" ::: "memory");      // per-wave LDS RAW fence
    v16bf pf = ldfrag(&Pl[wid][n16][0], h);           // P as A-operand (16x32)
    // O += P * V   (V^T rows give contiguous key-runs per output column d)
#pragma unroll
    for (int dt = 0; dt < 8; ++dt) {
      v16bf vf = ldfrag(Vp + (size_t)(dt*16 + n16) * LLEN + kbase, h);
#pragma unroll
      for (int r = 0; r < 8; ++r) ao[dt][r] *= alpha[r];
      ao[dt] = wmma_bf16(pf, vf, ao[dt]);
    }
  }

  // epilogue: normalize, write bf16 into [B*L, NH*HD] for the output projection
  float rinv[8];
#pragma unroll
  for (int r = 0; r < 8; ++r) rinv[r] = 1.0f / lrow[r];
#pragma unroll
  for (int dt = 0; dt < 8; ++dt)
#pragma unroll
    for (int r = 0; r < 8; ++r) {
      int q = q0 + r + 8*h;
      Ob[(size_t)(b * LLEN + q) * (NH * HDIM) + hq * HDIM + dt*16 + n16] =
          (__bf16)(ao[dt][r] * rinv[r]);
    }
}

// ---------------- host orchestration ----------------
static inline size_t alignup(size_t x) { return (x + 255) & ~(size_t)255; }

extern "C" void kernel_launch(void* const* d_in, const int* in_sizes, int n_in,
                              void* d_out, int out_size, void* d_ws, size_t ws_size,
                              hipStream_t stream) {
  const float* x  = (const float*)d_in[0];
  // d_in[1] = mask (unused; causal mask computed analytically)
  const float* wq = (const float*)d_in[2];
  const float* wk = (const float*)d_in[3];
  const float* wv = (const float*)d_in[4];
  const float* wo = (const float*)d_in[5];
  float* out = (float*)d_out;

  char* p = (char*)d_ws;
  size_t off = 0;
  auto take = [&](size_t bytes) { void* r = p + off; off = alignup(off + bytes); return r; };

  __bf16* xb   = (__bf16*)take((size_t)MROW * DIMX * 2);
  __bf16* wqT  = (__bf16*)take((size_t)DIMX * DIMX * 2);          // [4096][4096]
  __bf16* wkT  = (__bf16*)take((size_t)(NKV*HDIM) * DIMX * 2);    // [1024][4096]
  __bf16* wvT  = (__bf16*)take((size_t)(NKV*HDIM) * DIMX * 2);
  __bf16* woT  = (__bf16*)take((size_t)DIMX * DIMX * 2);
  float*  Qraw = (float*) take((size_t)MROW * (NH *HDIM) * 4);
  float*  Kraw = (float*) take((size_t)MROW * (NKV*HDIM) * 4);
  float*  Vraw = (float*) take((size_t)MROW * (NKV*HDIM) * 4);
  __bf16* Qb   = (__bf16*)take((size_t)BB * NH  * LLEN * HDIM * 2);
  __bf16* Kb   = (__bf16*)take((size_t)BB * NKV * LLEN * HDIM * 2);
  __bf16* Vt   = (__bf16*)take((size_t)BB * NKV * HDIM * LLEN * 2);
  __bf16* Ob   = (__bf16*)take((size_t)MROW * (NH*HDIM) * 2);
  (void)ws_size;

  // 1) convert x, convert+transpose weights to bf16
  {
    int n = MROW * DIMX;
    cvt_f32_bf16<<<(n + 255)/256, 256, 0, stream>>>(x, xb, n);
  }
  {
    dim3 blk(16, 16);
    cvt_transpose_bf16<<<dim3(DIMX/16,        DIMX/16), blk, 0, stream>>>(wq, wqT, DIMX, DIMX);
    cvt_transpose_bf16<<<dim3((NKV*HDIM)/16,  DIMX/16), blk, 0, stream>>>(wk, wkT, DIMX, NKV*HDIM);
    cvt_transpose_bf16<<<dim3((NKV*HDIM)/16,  DIMX/16), blk, 0, stream>>>(wv, wvT, DIMX, NKV*HDIM);
    cvt_transpose_bf16<<<dim3(DIMX/16,        DIMX/16), blk, 0, stream>>>(wo, woT, DIMX, DIMX);
  }
  // 2) QKV projections (WMMA GEMMs, TDM-staged)
  gemm_bf16_wmma<<<dim3((NH *HDIM)/128, MROW/128), 256, 0, stream>>>(xb, wqT, Qraw, MROW, NH *HDIM, DIMX);
  gemm_bf16_wmma<<<dim3((NKV*HDIM)/128, MROW/128), 256, 0, stream>>>(xb, wkT, Kraw, MROW, NKV*HDIM, DIMX);
  gemm_bf16_wmma<<<dim3((NKV*HDIM)/128, MROW/128), 256, 0, stream>>>(xb, wvT, Vraw, MROW, NKV*HDIM, DIMX);
  // 3) RoPE + relayout to head-major bf16; V transposed
  {
    int nq = BB * NH  * LLEN * (HDIM/2);
    int nk = BB * NKV * LLEN * (HDIM/2);
    int nv = BB * NKV * LLEN * HDIM;
    rope_relayout<<<(nq + 255)/256, 256, 0, stream>>>(Qraw, Qb, NH,  1);
    rope_relayout<<<(nk + 255)/256, 256, 0, stream>>>(Kraw, Kb, NKV, 1);
    v_relayout_t <<<(nv + 255)/256, 256, 0, stream>>>(Vraw, Vt);
  }
  // 4) causal flash attention
  flash_attn_wmma<<<dim3(LLEN/128, NH, BB), 256, 0, stream>>>(Qb, Kb, Vt, Ob);
  // 5) output projection -> f32 d_out
  gemm_bf16_wmma<<<dim3(DIMX/128, MROW/128), 256, 0, stream>>>(Ob, woT, out, MROW, DIMX, DIMX);
  (void)in_sizes; (void)n_in; (void)out_size;
}
